// BlockStnls_51668456570903
// MI455X (gfx1250) — matrix operand order repaired and checked by reading, not compiled, not run on hardware
//
#include <hip/hip_runtime.h>
#include <hip/hip_bf16.h>

// BlockStnls forward for MI455X (gfx1250), wave32 + WMMA f16->f32.
// v2: register-blocked GEMMs (4 row-tiles per wave, B fragment reuse x4).

typedef _Float16 f16;
typedef __attribute__((ext_vector_type(16))) _Float16 v16h;
typedef __attribute__((ext_vector_type(8)))  float    v8f;

#define C_    128
#define H_    128
#define W_    128
#define T_    4
#define HW_   16384
#define NPIX  65536
#define NOFF  147
#define SCALE 0.17677669529663689f   // 1/sqrt(32)

union U16h { v16h v; uint4 q[2]; };

#define WMMA16(a, b, c) \
  __builtin_amdgcn_wmma_f32_16x16x32_f16(false, (a), false, (b), (short)0, (c), false, false)

// A fragment (16x32 f16, M rows in LDS row-major with 128-half row stride):
// lane l: m = l&15, g = l>>4; halves = {kb+8g+0..7, kb+16+8g+0..7}  (ISA 7.12.2)
__device__ __forceinline__ v16h frag_a(const f16* __restrict__ base, int lane, int kb) {
  int m = lane & 15, g = lane >> 4;
  const f16* p = base + m * 128 + kb + g * 8;
  U16h u;
  u.q[0] = *(const uint4*)(p);
  u.q[1] = *(const uint4*)(p + 16);
  return u.v;
}

// B fragment (32x16 f16, K x N) from a row-major [n][ktot] f16 matrix:
// lane l: n = l&15, g = l>>4; halves = {kb+8g+0..7, kb+16+8g+0..7}
__device__ __forceinline__ v16h frag_b(const f16* __restrict__ bm, int lane,
                                       int n0, int ktot, int kb) {
  int n = lane & 15, g = lane >> 4;
  const f16* p = bm + (size_t)(n0 + n) * ktot + kb + g * 8;
  U16h u;
  u.q[0] = *(const uint4*)(p);
  u.q[1] = *(const uint4*)(p + 16);
  return u.v;
}

// ---------------- weight packing ----------------
__global__ __launch_bounds__(256) void pack_proj_kernel(
    const float* __restrict__ wq, const float* __restrict__ wk,
    const float* __restrict__ wv, const float* __restrict__ wo,
    f16* __restrict__ wqh, f16* __restrict__ wkh,
    f16* __restrict__ wvh, f16* __restrict__ woh) {
  int i = blockIdx.x * 256 + threadIdx.x;
  if (i < C_ * C_) {
    wqh[i] = (f16)wq[i]; wkh[i] = (f16)wk[i];
    wvh[i] = (f16)wv[i]; woh[i] = (f16)wo[i];
  }
}

__global__ __launch_bounds__(256) void pack_conv_kernel(
    const float* __restrict__ w1, const float* __restrict__ w2,
    f16* __restrict__ w1h, f16* __restrict__ w2h) {
  int i = blockIdx.x * 256 + threadIdx.x;
  if (i < C_ * C_ * 9) {
    int o = i / 1152, rem = i % 1152, kidx = rem >> 7, c = rem & 127;
    size_t src = ((size_t)o * C_ + c) * 9 + kidx;      // OIHW, kidx = kh*3+kw
    w1h[i] = (f16)w1[src];
    w2h[i] = (f16)w2[src];
  }
}

// ---------------- layernorm over C (per pixel, one wave) ----------------
__global__ __launch_bounds__(256) void ln_kernel(
    const float* __restrict__ src,   // [T][C][H][W]
    const float* __restrict__ g, const float* __restrict__ b,
    f16* __restrict__ outh) {        // [n][C] pixel-major f16
  int n = blockIdx.x * 8 + (threadIdx.x >> 5);
  int lane = threadIdx.x & 31;
  int t = n >> 14, hw = n & (HW_ - 1);
  const float* base = src + (size_t)t * C_ * HW_ + hw;
  float x[4]; float s = 0.f, s2 = 0.f;
#pragma unroll
  for (int i = 0; i < 4; ++i) {
    x[i] = base[(size_t)(lane * 4 + i) * HW_];
    s += x[i]; s2 += x[i] * x[i];
  }
#pragma unroll
  for (int m = 1; m < 32; m <<= 1) {
    s  += __shfl_xor(s,  m, 32);
    s2 += __shfl_xor(s2, m, 32);
  }
  float mu  = s  * (1.f / 128.f);
  float var = s2 * (1.f / 128.f) - mu * mu;
  float inv = rsqrtf(var + 1e-6f);
#pragma unroll
  for (int i = 0; i < 4; ++i) {
    int c = lane * 4 + i;
    outh[(size_t)n * C_ + c] = (f16)((x[i] - mu) * inv * g[c] + b[c]);
  }
}

// ---------------- QKV projection GEMM (WMMA, 64-row blocks) ----------------
__global__ __launch_bounds__(256) void gemm_qkv_kernel(
    const f16* __restrict__ xh,
    const f16* __restrict__ wqh, const f16* __restrict__ wkh, const f16* __restrict__ wvh,
    float* __restrict__ qf, float* __restrict__ kf, float* __restrict__ vf) {
  __shared__ __align__(16) f16 As[64 * 128];
  int nbase = blockIdx.x * 64;
  int mat = blockIdx.y;
  int tid = threadIdx.x;
  {
    const uint4* src = (const uint4*)(xh + (size_t)nbase * C_);
    uint4* dst = (uint4*)As;
#pragma unroll
    for (int i = 0; i < 4; ++i) dst[tid + i * 256] = src[tid + i * 256];
  }
  __syncthreads();
  int lane = tid & 31, wv = tid >> 5;
  int colbase = wv * 16;
  const f16* B = (mat == 0) ? wqh : (mat == 1) ? wkh : wvh;
  float*     O = (mat == 0) ? qf  : (mat == 1) ? kf  : vf;
  __builtin_prefetch(B + (size_t)colbase * C_, 0, 0);
  v8f acc[4] = {};
#pragma unroll
  for (int kb = 0; kb < 128; kb += 32) {
    v16h b = frag_b(B, lane, colbase, 128, kb);
#pragma unroll
    for (int rt = 0; rt < 4; ++rt) {
      v16h a = frag_a(As + rt * 16 * 128, lane, kb);
      acc[rt] = WMMA16(a, b, acc[rt]);
    }
  }
  int m0 = (lane >> 4) * 8, nc = lane & 15;
#pragma unroll
  for (int rt = 0; rt < 4; ++rt)
#pragma unroll
    for (int j = 0; j < 8; ++j)
      O[(size_t)(nbase + rt * 16 + m0 + j) * C_ + colbase + nc] = acc[rt][j];
}

// ---------------- flow-warped non-local attention (one wave per pixel) ----------------
__global__ __launch_bounds__(256) void attn_kernel(
    const float* __restrict__ qf, const float* __restrict__ kf,
    const float* __restrict__ vf, const float* __restrict__ flows,
    f16* __restrict__ attnh) {
  __shared__ float sc[8][4 * NOFF];
  __shared__ int   fl[8][NOFF];
  int wv = threadIdx.x >> 5, lane = threadIdx.x & 31;
  int n = blockIdx.x * 8 + wv;
  int t = n >> 14, h = (n >> 7) & 127, w = n & 127;
  int head = lane >> 3;
  float q[4];
#pragma unroll
  for (int i = 0; i < 4; ++i) q[i] = qf[(size_t)n * C_ + lane * 4 + i];
  float fh = flows[(((size_t)t * 2 + 0) * H_ + h) * W_ + w];
  float fw = flows[(((size_t)t * 2 + 1) * H_ + h) * W_ + w];
  float mx = -1e30f;
  for (int off = 0; off < NOFF; ++off) {
    int dt = off / 49 - 1, rem = off % 49, dh = rem / 7 - 3, dw = rem % 7 - 3;
    int ti = min(max(t + dt, 0), T_ - 1);
    int oh = (int)rintf(fh * (float)dt);
    int ow = (int)rintf(fw * (float)dt);
    int hi = min(max(h + oh + dh, 0), H_ - 1);
    int wi = min(max(w + ow + dw, 0), W_ - 1);
    int flat = (ti * H_ + hi) * W_ + wi;
    const float* kp = kf + (size_t)flat * C_ + lane * 4;
    float p = q[0] * kp[0] + q[1] * kp[1] + q[2] * kp[2] + q[3] * kp[3];
#pragma unroll
    for (int m = 1; m < 8; m <<= 1) p += __shfl_xor(p, m, 32);  // reduce within head
    p *= SCALE;
    if ((lane & 7) == 0) sc[wv][head * NOFF + off] = p;
    if (lane == 0) fl[wv][off] = flat;
    mx = fmaxf(mx, p);
  }
  __syncthreads();
  float sum = 0.f;
  for (int off = lane & 7; off < NOFF; off += 8) {
    float e = __expf(sc[wv][head * NOFF + off] - mx);
    sc[wv][head * NOFF + off] = e;
    sum += e;
  }
#pragma unroll
  for (int m = 1; m < 8; m <<= 1) sum += __shfl_xor(sum, m, 32);
  float inv = 1.f / sum;
  __syncthreads();
  float a0 = 0.f, a1 = 0.f, a2 = 0.f, a3 = 0.f;
  for (int off = 0; off < NOFF; ++off) {
    float wg = sc[wv][head * NOFF + off] * inv;
    const float* vp = vf + (size_t)fl[wv][off] * C_ + lane * 4;
    a0 += wg * vp[0]; a1 += wg * vp[1]; a2 += wg * vp[2]; a3 += wg * vp[3];
  }
  f16* op = attnh + (size_t)n * C_ + lane * 4;
  op[0] = (f16)a0; op[1] = (f16)a1; op[2] = (f16)a2; op[3] = (f16)a3;
}

// ---------------- output projection + residual (WMMA, 64-row blocks) ----------------
__global__ __launch_bounds__(256) void gemm_o_kernel(
    const f16* __restrict__ attnh, const f16* __restrict__ woh,
    const float* __restrict__ vid, float* __restrict__ vid1) {
  __shared__ __align__(16) f16 As[64 * 128];
  int nbase = blockIdx.x * 64;
  int tid = threadIdx.x;
  {
    const uint4* src = (const uint4*)(attnh + (size_t)nbase * C_);
    uint4* dst = (uint4*)As;
#pragma unroll
    for (int i = 0; i < 4; ++i) dst[tid + i * 256] = src[tid + i * 256];
  }
  __syncthreads();
  int lane = tid & 31, wv = tid >> 5;
  int colbase = wv * 16;
  __builtin_prefetch(woh + (size_t)colbase * C_, 0, 0);
  v8f acc[4] = {};
#pragma unroll
  for (int kb = 0; kb < 128; kb += 32) {
    v16h b = frag_b(woh, lane, colbase, 128, kb);
#pragma unroll
    for (int rt = 0; rt < 4; ++rt) {
      v16h a = frag_a(As + rt * 16 * 128, lane, kb);
      acc[rt] = WMMA16(a, b, acc[rt]);
    }
  }
  int t = nbase >> 14, h = (nbase >> 7) & 127, w0 = nbase & 127;
  int m0 = (lane >> 4) * 8, o = colbase + (lane & 15);
#pragma unroll
  for (int rt = 0; rt < 4; ++rt) {
#pragma unroll
    for (int j = 0; j < 8; ++j) {
      size_t idx = (((size_t)t * C_ + o) * H_ + h) * W_ + (w0 + rt * 16 + m0 + j);
      vid1[idx] = vid[idx] + acc[rt][j];
    }
  }
}

// ---------------- 3x3 conv as implicit GEMM (WMMA, 64-pixel blocks) ----------------
__global__ __launch_bounds__(256) void conv_kernel(
    const f16* __restrict__ inh,     // [n][C] pixel-major f16
    const f16* __restrict__ wpack,   // [o][kidx*128+c] f16
    const float* __restrict__ bias,
    f16* __restrict__ outh,                                   // mode 0: gelu(conv) as f16
    const float* __restrict__ vid1, const f16* __restrict__ yh,
    float* __restrict__ out,                                  // mode 1: final fused output
    int mode) {
  __shared__ __align__(16) f16 Ss[3 * 66 * 128];  // 3 rows x 66 cols x 128 ch halo tile
  int nbase = blockIdx.x * 64;
  int t = nbase >> 14, h = (nbase >> 7) & 127, w0 = nbase & 127;
  int tid = threadIdx.x;
  for (int i = tid; i < 3 * 66 * 16; i += 256) {  // 198 pixels x 16 chunks of 8 halves
    int pix = i >> 4, ch = i & 15;
    int r = pix / 66, cc = pix % 66;
    int hh = h - 1 + r, ww = w0 - 1 + cc;
    uint4 val = {0u, 0u, 0u, 0u};
    if (hh >= 0 && hh < H_ && ww >= 0 && ww < W_) {
      size_t np = (size_t)t * HW_ + hh * W_ + ww;
      val = *(const uint4*)(inh + np * C_ + ch * 8);
    }
    *(uint4*)(Ss + pix * 128 + ch * 8) = val;
  }
  __syncthreads();
  int lane = tid & 31, wv = tid >> 5;
  int colbase = wv * 16;
  __builtin_prefetch(wpack + (size_t)colbase * 1152, 0, 0);
  v8f acc[4] = {};
  for (int kidx = 0; kidx < 9; ++kidx) {
    int kh = kidx / 3, kw = kidx % 3;
    const f16* abase = Ss + (kh * 66 + kw) * 128;  // frag_a adds m*128 -> pixel (m+kw)
#pragma unroll
    for (int cb = 0; cb < 128; cb += 32) {
      v16h b = frag_b(wpack, lane, colbase, 1152, kidx * 128 + cb);
#pragma unroll
      for (int rt = 0; rt < 4; ++rt) {
        v16h a = frag_a(abase + rt * 16 * 128, lane, cb);
        acc[rt] = WMMA16(a, b, acc[rt]);
      }
    }
  }
  int m0 = (lane >> 4) * 8, o = colbase + (lane & 15);
  float bo = bias[o];
  if (mode == 0) {
#pragma unroll
    for (int rt = 0; rt < 4; ++rt) {
#pragma unroll
      for (int j = 0; j < 8; ++j) {
        float x = acc[rt][j] + bo;
        float g = 0.5f * x * (1.f + tanhf(0.7978845608028654f * (x + 0.044715f * x * x * x)));
        outh[(size_t)(nbase + rt * 16 + m0 + j) * C_ + o] = (f16)g;
      }
    }
  } else {
#pragma unroll
    for (int rt = 0; rt < 4; ++rt) {
#pragma unroll
      for (int j = 0; j < 8; ++j) {
        int mm = rt * 16 + m0 + j;
        size_t pidx = (((size_t)t * C_ + o) * H_ + h) * W_ + (w0 + mm);
        float yv = (float)yh[(size_t)(nbase + mm) * C_ + o];
        out[pidx] = vid1[pidx] + yv + acc[rt][j] + bo;  // vid1 + (yb + r)
      }
    }
  }
}

// ---------------- host launch ----------------
extern "C" void kernel_launch(void* const* d_in, const int* in_sizes, int n_in,
                              void* d_out, int out_size, void* d_ws, size_t ws_size,
                              hipStream_t stream) {
  const float* vid   = (const float*)d_in[0];
  const float* flows = (const float*)d_in[1];
  const float* g0    = (const float*)d_in[2];
  const float* b0    = (const float*)d_in[3];
  const float* Wq    = (const float*)d_in[4];
  const float* Wk    = (const float*)d_in[5];
  const float* Wv    = (const float*)d_in[6];
  const float* Wo    = (const float*)d_in[7];
  const float* g1    = (const float*)d_in[8];
  const float* b1    = (const float*)d_in[9];
  const float* w_r1  = (const float*)d_in[10];
  const float* b_r1  = (const float*)d_in[11];
  const float* w_r2  = (const float*)d_in[12];
  const float* b_r2  = (const float*)d_in[13];
  float* out = (float*)d_out;

  char* ws = (char*)d_ws;
  const size_t MB = 1ull << 20;
  // region A (16 MiB): xh -> attnh -> yh (sequential reuse)
  f16*   xh    = (f16*)(ws + 0);
  f16*   attnh = (f16*)(ws + 0);
  f16*   yh    = (f16*)(ws + 0);
  // region B (32 MiB): qf -> r1h
  float* qf    = (float*)(ws + 16 * MB);
  f16*   r1h   = (f16*)  (ws + 16 * MB);
  float* kf    = (float*)(ws + 48 * MB);
  float* vf    = (float*)(ws + 80 * MB);
  float* vid1  = (float*)(ws + 112 * MB);
  f16*   wqh   = (f16*)(ws + 144 * MB);
  f16*   wkh   = wqh + 16384;
  f16*   wvh   = wkh + 16384;
  f16*   woh   = wvh + 16384;
  f16*   w1h   = woh + 16384;
  f16*   w2h   = w1h + 147456;

  pack_proj_kernel<<<64, 256, 0, stream>>>(Wq, Wk, Wv, Wo, wqh, wkh, wvh, woh);
  pack_conv_kernel<<<576, 256, 0, stream>>>(w_r1, w_r2, w1h, w2h);

  ln_kernel<<<NPIX / 8, 256, 0, stream>>>(vid, g0, b0, xh);
  gemm_qkv_kernel<<<dim3(NPIX / 64, 3), 256, 0, stream>>>(xh, wqh, wkh, wvh, qf, kf, vf);
  attn_kernel<<<NPIX / 8, 256, 0, stream>>>(qf, kf, vf, flows, attnh);
  gemm_o_kernel<<<NPIX / 64, 256, 0, stream>>>(attnh, woh, vid, vid1);
  ln_kernel<<<NPIX / 8, 256, 0, stream>>>(vid1, g1, b1, yh);
  conv_kernel<<<NPIX / 64, 256, 0, stream>>>(yh, w1h, b_r1, r1h, nullptr, nullptr, nullptr, 0);
  conv_kernel<<<NPIX / 64, 256, 0, stream>>>(r1h, w2h, b_r2, nullptr, vid1, yh, out, 1);
}
